// Encoder_45286135169498
// MI455X (gfx1250) — matrix operand order repaired and checked by reading, not compiled, not run on hardware
//
#include <hip/hip_runtime.h>
#include <hip/hip_bf16.h>

#define D_PRE   768
#define DMODEL  512
#define NHEAD   4
#define DH      128
#define NEXP    8
#define DFF     2048
#define NB      32
#define STOK    64
#define SGEO    12
#define SCAT    76
#define NTOK    (NB*STOK)   /* 2048 */
#define NCAT    (NB*SCAT)   /* 2432 */
#define CAP_TOK 512
#define CAP_GEO 608

#define BM 128
#define BN 64
#define BK 32
#define AST 18   /* Ap row stride in dwords: even -> 8B-aligned fragment chunks */
#define BST 18   /* Bp col stride in dwords */

typedef __attribute__((ext_vector_type(16))) __bf16 v16bf;
typedef __attribute__((ext_vector_type(8)))  float  v8f;
typedef __attribute__((ext_vector_type(8)))  unsigned int v8u;

__device__ __forceinline__ unsigned short f2bf(float f) {
  unsigned int u = __float_as_uint(f);
  u += 0x7FFFu + ((u >> 16) & 1u);      // round to nearest even
  return (unsigned short)(u >> 16);
}
__device__ __forceinline__ unsigned int pack2bf(float a, float b) {
  return (unsigned int)f2bf(a) | ((unsigned int)f2bf(b) << 16);
}
__device__ __forceinline__ float gelu_tanh(float x) {
  float x3 = x * x * x;
  return 0.5f * x * (1.0f + tanhf(0.7978845608028654f * (x + 0.044715f * x3)));
}

// ---------------------------------------------------------------------------
// WMMA GEMM: C[M,N] = act(A[M,K] @ B[K,N] + bias) + R   (batched via grid.z)
// ACT: 0=none 1=leaky(0.01) 2=gelu(tanh) 3=relu   (compile-time)
// Requirements (met by every call site): K % 32 == 0, K >= 64.
// Out-of-range rows/cols handled by CLAMPING load indices (junk only feeds
// outputs the store guard drops) -> unconditional coalesced loads.
// LDS holds bf16 pairs packed as dwords in fragment order (8B aligned);
// double-buffered ping-pong: one barrier per K-step, next-tile global loads
// issued before the MMA phase; all fragments fetched before the 4 WMMAs so
// DS waits can be progressive and WMMAs go back-to-back.
// ---------------------------------------------------------------------------
template <int ACT>
__global__ __launch_bounds__(256) void gemm_kernel(
    const float* __restrict__ A, const float* __restrict__ B,
    const float* __restrict__ bias, const float* __restrict__ R,
    float* __restrict__ C,
    int M, int N, int K, int lda, int ldb, int ldc, int ldr,
    long long sA, long long sB, long long sC, long long sBias, long long sR)
{
  __shared__ unsigned int Ap[2][BM * AST];   // 2*128*18*4 = 18432 B
  __shared__ unsigned int Bp[2][BN * BST];   // 2* 64*18*4 =  9216 B

  int z = blockIdx.z;
  A += (long long)z * sA;
  B += (long long)z * sB;
  C += (long long)z * sC;
  const float* biasz = bias ? bias + (long long)z * sBias : nullptr;
  const float* Rz    = R    ? R    + (long long)z * sR    : nullptr;

  int m0 = blockIdx.y * BM;
  int n0 = blockIdx.x * BN;
  int tid  = threadIdx.x;
  int lane = tid & 31;
  int w    = tid >> 5;           // wave 0..7 -> rows w*16 .. w*16+15
  int hh   = lane >> 4;          // lane half
  int rr   = lane & 15;          // A row / B,D col within 16

  v8f acc[4] = {v8f{}, v8f{}, v8f{}, v8f{}};

  auto load_tiles = [&](int k0, int buf) {
    // A tile: 128 rows x 16 kpairs; slot i: r = i>>4, kp = i&15
#pragma unroll
    for (int t = 0; t < (BM * 16) / 256; ++t) {
      int i = tid + t * 256;
      int r = i >> 4, kp = i & 15;
      int gm = m0 + r; if (gm >= M) gm = M - 1;          // clamp
      const float2 v = *(const float2*)(A + (long long)gm * lda + k0 + 2 * kp);
      Ap[buf][r * AST + kp] = pack2bf(v.x, v.y);
    }
    // B tile: 16 kpairs x 64 cols; slot i: kp = i>>6, c = i&63
#pragma unroll
    for (int t = 0; t < (16 * BN) / 256; ++t) {
      int i = tid + t * 256;
      int kp = i >> 6, c = i & 63;
      int gn = n0 + c; if (gn >= N) gn = N - 1;          // clamp
      const float* src = B + (long long)(k0 + 2 * kp) * ldb + gn;
      Bp[buf][c * BST + kp] = pack2bf(src[0], src[ldb]);
    }
  };

  auto compute = [&](int buf) {
    // A fragment: dwords {8h..8h+7} (contiguous k = 16h + s, matches B)
    v8u af;
    {
      const unsigned int* ar = &Ap[buf][(w * 16 + rr) * AST];
#pragma unroll
      for (int d = 0; d < 4; ++d) af[d] = ar[8 * hh + d];
#pragma unroll
      for (int d = 0; d < 4; ++d) af[4 + d] = ar[8 * hh + 4 + d];
    }
    // All 4 B fragments first (cols q*16 + rr): dword d -> kpair 8h + d
    v8u bfr0, bfr1, bfr2, bfr3;
    {
      const unsigned int* br0 = &Bp[buf][( 0 + rr) * BST + 8 * hh];
      const unsigned int* br1 = &Bp[buf][(16 + rr) * BST + 8 * hh];
      const unsigned int* br2 = &Bp[buf][(32 + rr) * BST + 8 * hh];
      const unsigned int* br3 = &Bp[buf][(48 + rr) * BST + 8 * hh];
#pragma unroll
      for (int d = 0; d < 8; ++d) bfr0[d] = br0[d];
#pragma unroll
      for (int d = 0; d < 8; ++d) bfr1[d] = br1[d];
#pragma unroll
      for (int d = 0; d < 8; ++d) bfr2[d] = br2[d];
#pragma unroll
      for (int d = 0; d < 8; ++d) bfr3[d] = br3[d];
    }
    v16bf a = __builtin_bit_cast(v16bf, af);
    acc[0] = __builtin_amdgcn_wmma_f32_16x16x32_bf16(false, a, false,
               __builtin_bit_cast(v16bf, bfr0), (short)0, acc[0], false, false);
    acc[1] = __builtin_amdgcn_wmma_f32_16x16x32_bf16(false, a, false,
               __builtin_bit_cast(v16bf, bfr1), (short)0, acc[1], false, false);
    acc[2] = __builtin_amdgcn_wmma_f32_16x16x32_bf16(false, a, false,
               __builtin_bit_cast(v16bf, bfr2), (short)0, acc[2], false, false);
    acc[3] = __builtin_amdgcn_wmma_f32_16x16x32_bf16(false, a, false,
               __builtin_bit_cast(v16bf, bfr3), (short)0, acc[3], false, false);
  };

  int nk = K / BK;
  load_tiles(0, 0);
  int cur = 0;
  for (int kt = 0; kt + 1 < nk; ++kt) {      // steady state: branch-free body
    __syncthreads();
    load_tiles((kt + 1) * BK, cur ^ 1);
    compute(cur);
    cur ^= 1;
  }
  __syncthreads();                            // peeled last iteration
  compute(cur);

  // D layout: lane col = rr, VGPR i -> row i + 8*hh
#pragma unroll
  for (int q = 0; q < 4; ++q) {
    int col = n0 + q * 16 + rr;
    if (col >= N) continue;
    float bv = biasz ? biasz[col] : 0.0f;
#pragma unroll
    for (int i = 0; i < 8; ++i) {
      int row = m0 + w * 16 + i + 8 * hh;
      if (row >= M) continue;
      float v = acc[q][i] + bv;
      if (ACT == 1)      v = (v >= 0.f) ? v : 0.01f * v;
      else if (ACT == 2) v = gelu_tanh(v);
      else if (ACT == 3) v = fmaxf(v, 0.f);
      if (Rz) v += Rz[(long long)row * ldr + col];
      C[(long long)row * ldc + col] = v;
    }
  }
}

// ---------------------------------------------------------------------------
// Y = LayerNorm(X + A2) * g + b   (A2 may be null). One block per row.
// ---------------------------------------------------------------------------
__global__ __launch_bounds__(256) void addln_kernel(
    const float* __restrict__ X, const float* __restrict__ A2,
    const float* __restrict__ g, const float* __restrict__ b,
    float* __restrict__ Y, int dim)
{
  __shared__ float buf[768];
  __shared__ float red0[256];
  __shared__ float red1[256];
  int row = blockIdx.x;
  const float* x = X + (long long)row * dim;
  const float* a = A2 ? A2 + (long long)row * dim : nullptr;
  float s = 0.f, s2 = 0.f;
  for (int i = threadIdx.x; i < dim; i += 256) {
    float v = x[i] + (a ? a[i] : 0.f);
    buf[i] = v; s += v; s2 += v * v;
  }
  red0[threadIdx.x] = s; red1[threadIdx.x] = s2;
  __syncthreads();
  for (int off = 128; off > 0; off >>= 1) {
    if (threadIdx.x < off) {
      red0[threadIdx.x] += red0[threadIdx.x + off];
      red1[threadIdx.x] += red1[threadIdx.x + off];
    }
    __syncthreads();
  }
  float m   = red0[0] / dim;
  float var = red1[0] / dim - m * m;
  float inv = rsqrtf(var + 1e-5f);
  for (int i = threadIdx.x; i < dim; i += 256)
    Y[(long long)row * dim + i] = (buf[i] - m) * inv * g[i] + b[i];
}

// ---------------------------------------------------------------------------
// MoE routing: softmax over 8 logits, top-2, thresholds {0,0.2}, normalize;
// accumulate density (balance) and logsumexp^2 (zloss) via atomics.
// ---------------------------------------------------------------------------
__global__ void route_kernel(const float* __restrict__ logits,
                             float* __restrict__ gval, int* __restrict__ gidx,
                             int* __restrict__ keep, float* __restrict__ lossacc,
                             int N)
{
  int r = blockIdx.x * blockDim.x + threadIdx.x;
  if (r >= N) return;
  float l[NEXP], p[NEXP];
  float mx = -1e30f;
  for (int e = 0; e < NEXP; ++e) { l[e] = logits[r * NEXP + e]; mx = fmaxf(mx, l[e]); }
  float se = 0.f;
  for (int e = 0; e < NEXP; ++e) { p[e] = expf(l[e] - mx); se += p[e]; }
  float inv = 1.0f / se;
  for (int e = 0; e < NEXP; ++e) p[e] *= inv;
  int i1 = 0; float v1 = p[0];
  for (int e = 1; e < NEXP; ++e) if (p[e] > v1) { v1 = p[e]; i1 = e; }
  int i2 = (i1 == 0) ? 1 : 0; float v2 = p[i2];
  for (int e = 0; e < NEXP; ++e) { if (e == i1 || e == i2) continue; if (p[e] > v2) { v2 = p[e]; i2 = e; } }
  int k0 = (v1 >= 0.0f) ? 1 : 0;
  int k1 = (v2 >= 0.2f) ? 1 : 0;
  float gv0 = v1 * k0, gv1 = v2 * k1;
  float den = gv0 + gv1 + 1e-9f;
  gval[r * 2] = gv0 / den; gval[r * 2 + 1] = gv1 / den;
  gidx[r * 2] = i1; gidx[r * 2 + 1] = i2;
  keep[r * 2] = k0; keep[r * 2 + 1] = k1;
  float invN = 1.0f / (float)N;
  for (int e = 0; e < NEXP; ++e) atomicAdd(&lossacc[e], p[e] * invN);
  float lse = mx + logf(se);
  atomicAdd(&lossacc[NEXP], lse * lse * invN);
}

// Per-expert deterministic slot assignment: wave w scans flat (token,slot)
// order with wave32 ballot prefix sums (matches reference cumsum semantics).
__global__ void scan_kernel(const int* __restrict__ gidx, const int* __restrict__ keep,
                            int* __restrict__ pos, int NF)
{
  int lane = threadIdx.x & 31;
  int e    = threadIdx.x >> 5;   // 8 waves = 8 experts
  int base = 0;
  for (int f0 = 0; f0 < NF; f0 += 32) {
    int f = f0 + lane;
    bool match = (f < NF) && (gidx[f] == e) && (keep[f] != 0);
    unsigned m = (unsigned)__ballot(match);
    if (match) pos[f] = base + __popc(m & ((1u << lane) - 1u));
    base += __popc(m);
  }
}

__global__ __launch_bounds__(256) void scatter_kernel(
    const float* __restrict__ X, const int* __restrict__ gidx,
    const int* __restrict__ keep, const int* __restrict__ pos,
    float* __restrict__ buf, int cap)
{
  int f = blockIdx.x;
  if (!keep[f]) return;
  int p = pos[f];
  if (p >= cap) return;
  int e = gidx[f];
  int r = f >> 1;
  const float* src = X + (long long)r * DMODEL;
  float* dst = buf + ((long long)e * cap + p) * DMODEL;
  for (int i = threadIdx.x; i < DMODEL; i += 256) dst[i] = src[i];
}

__global__ __launch_bounds__(256) void combine_kernel(
    float* __restrict__ X, const float* __restrict__ obuf,
    const float* __restrict__ gval, const int* __restrict__ gidx,
    const int* __restrict__ keep, const int* __restrict__ pos, int cap)
{
  int r = blockIdx.x;
  int f = r * 2;
  float w0 = 0.f, w1 = 0.f;
  const float* s0 = nullptr; const float* s1 = nullptr;
  if (keep[f]     && pos[f]     < cap) { w0 = gval[f];     s0 = obuf + ((long long)gidx[f]     * cap + pos[f])     * DMODEL; }
  if (keep[f + 1] && pos[f + 1] < cap) { w1 = gval[f + 1]; s1 = obuf + ((long long)gidx[f + 1] * cap + pos[f + 1]) * DMODEL; }
  float* x = X + (long long)r * DMODEL;
  for (int i = threadIdx.x; i < DMODEL; i += 256) {
    float v = x[i];
    if (s0) v += w0 * s0[i];
    if (s1) v += w1 * s1[i];
    x[i] = v;
  }
}

// GLU in place: rows of width 2*DFF; H[:, :DFF] *= gelu(H[:, DFF:])
__global__ void glu_kernel(float* __restrict__ H, long long rows)
{
  long long idx = (long long)blockIdx.x * blockDim.x + threadIdx.x;
  long long total = rows * (long long)DFF;
  if (idx >= total) return;
  long long r = idx / DFF; int j = (int)(idx % DFF);
  float* row = H + r * (2LL * DFF);
  row[j] = row[j] * gelu_tanh(row[j + DFF]);
}

// cat = concat(tok, geo_emb[geo_code]) * sqrt(D) + PE
__global__ void concat_pe_kernel(const float* __restrict__ tok,
                                 const float* __restrict__ geo_emb,
                                 const int* __restrict__ geo_code,
                                 float* __restrict__ X)
{
  int idx = blockIdx.x * blockDim.x + threadIdx.x;
  if (idx >= NCAT * DMODEL) return;
  int c  = idx % DMODEL;
  int bs = idx / DMODEL;
  int s  = bs % SCAT, b = bs / SCAT;
  float v;
  if (s < STOK) v = tok[((long long)b * STOK + s) * DMODEL + c];
  else {
    int code = geo_code[b * SGEO + (s - STOK)];
    v = geo_emb[(long long)code * DMODEL + c];
  }
  int i2 = c >> 1;
  float freq = expf((float)(2 * i2) * (-9.210340371976184f / 512.0f));
  float ang  = (float)s * freq;
  float pe   = (c & 1) ? cosf(ang) : sinf(ang);
  X[idx] = v * 22.62741699796952f + pe;   // sqrt(512)
}

// One block per (b, head). K tile in LDS; V reads coalesced from global.
__global__ __launch_bounds__(128) void attn_kernel(
    const float* __restrict__ qkv, const int* __restrict__ token_mask,
    float* __restrict__ O)
{
  __shared__ float Ks[SCAT][DH + 4];
  __shared__ float qv[DH];
  __shared__ float sc[SCAT];
  int bh = blockIdx.x;
  int b = bh / NHEAD, h = bh % NHEAD;
  int tid = threadIdx.x;
  const float* base = qkv + (long long)b * SCAT * (3 * DMODEL);
  for (int i = tid; i < SCAT * DH; i += 128) {
    int r = i / DH, d = i % DH;
    Ks[r][d] = base[(long long)r * (3 * DMODEL) + DMODEL + h * DH + d];
  }
  __syncthreads();
  for (int sq = 0; sq < SCAT; ++sq) {
    qv[tid] = base[(long long)sq * (3 * DMODEL) + h * DH + tid];
    __syncthreads();
    if (tid < SCAT) {
      bool masked = (tid < STOK) && (token_mask[b * STOK + tid] == 0);
      float s;
      if (masked) s = -1e9f;
      else {
        s = 0.f;
        for (int d = 0; d < DH; ++d) s += qv[d] * Ks[tid][d];
        s *= 0.08838834764831845f;     // 1/sqrt(128)
      }
      sc[tid] = s;
    }
    __syncthreads();
    if (tid == 0) {
      float mx = -1e30f;
      for (int k = 0; k < SCAT; ++k) mx = fmaxf(mx, sc[k]);
      float se = 0.f;
      for (int k = 0; k < SCAT; ++k) { sc[k] = expf(sc[k] - mx); se += sc[k]; }
      float inv = 1.f / se;
      for (int k = 0; k < SCAT; ++k) sc[k] *= inv;
    }
    __syncthreads();
    float o = 0.f;
    for (int k = 0; k < SCAT; ++k)
      o += sc[k] * base[(long long)k * (3 * DMODEL) + 2 * DMODEL + h * DH + tid];
    O[((long long)b * SCAT + sq) * DMODEL + h * DH + tid] = o;
    __syncthreads();
  }
}

__global__ void slice_geo_kernel(const float* __restrict__ X, float* __restrict__ out)
{
  int idx = blockIdx.x * blockDim.x + threadIdx.x;
  if (idx >= NB * SGEO * DMODEL) return;
  int c  = idx % DMODEL;
  int bj = idx / DMODEL;
  int j = bj % SGEO, b = bj / SGEO;
  out[idx] = X[((long long)b * SCAT + STOK + j) * DMODEL + c];
}

__global__ void aux_finalize_kernel(const float* __restrict__ lossacc, float* __restrict__ auxout)
{
  if (threadIdx.x == 0 && blockIdx.x == 0) {
    float s = 0.f;
    for (int e = 0; e < NEXP; ++e) s += lossacc[e] * lossacc[e];
    atomicAdd(auxout, s * (float)NEXP * 0.01f + lossacc[NEXP] * 0.001f);
  }
}

// ---------------------------------------------------------------------------
extern "C" void kernel_launch(void* const* d_in, const int* in_sizes, int n_in,
                              void* d_out, int out_size, void* d_ws, size_t ws_size,
                              hipStream_t stream)
{
  (void)in_sizes; (void)n_in; (void)out_size; (void)ws_size;

  const float* text_emb  = (const float*)d_in[0];
  const float* token_emb = (const float*)d_in[1];
  const int*   token_mask= (const int*)d_in[2];
  const int*   geo_code  = (const int*)d_in[3];
  const float* const* P  = (const float* const*)(d_in + 4);
  // P[0..5]  text {ln_g,ln_b,w1,b1,w2,b2}; P[6..11] token
  // P[12..28] tokmoe: ffb(6), ffa(6), gate, ew1, eb1, ew2, eb2
  // P[29..45] geomoe; P[46 + 12i ..] tl{i}; P[94] geo_emb

  float* W  = (float*)d_ws;
  float* RA = W;                                   // 2432 x 512
  float* RB = RA + (size_t)NCAT * DMODEL;          // 2432 x 2048
  float* RC = RB + (size_t)NCAT * DFF;             // 2432 x 512
  float* RD = RC + (size_t)NCAT * DMODEL;          // 2048 x 512 (tok)
  float* RE = RD + (size_t)NTOK * DMODEL;          // 8 x 608 x 512 (expert in)
  float* RF = RE + (size_t)NEXP * CAP_GEO * DMODEL;// 8 x 608 x 4096 (expert mid)
  float* RG = RF + (size_t)NEXP * CAP_GEO * 2 * DFF;// 8 x 608 x 512 (expert out)
  float* LG = RG + (size_t)NEXP * CAP_GEO * DMODEL;// logits 2432 x 8
  float* GV = LG + (size_t)NCAT * NEXP;            // gate vals 2432 x 2
  int*   GI = (int*)(GV + (size_t)NCAT * 2);
  int*   KP = GI + (size_t)NCAT * 2;
  int*   PS = KP + (size_t)NCAT * 2;
  float* LA = (float*)(PS + (size_t)NCAT * 2);     // 9 loss accumulators

  float* out_text = (float*)d_out;                           // 32*512
  float* out_tok  = out_text + (size_t)NB * DMODEL;          // 2048*512
  float* out_geo  = out_tok  + (size_t)NTOK * DMODEL;        // 32*12*512
  float* out_aux  = out_geo  + (size_t)NB * SGEO * DMODEL;   // 1

  auto GEMM = [&](const float* A, const float* Bm, const float* bias, const float* R,
                  float* Cm, int M, int N, int K, int lda, int ldb, int ldc, int ldr,
                  long long sA, long long sB, long long sC, long long sBias, long long sR,
                  int batch, int act) {
    dim3 grid((N + BN - 1) / BN, (M + BM - 1) / BM, batch);
    switch (act) {
      case 1: gemm_kernel<1><<<grid, 256, 0, stream>>>(A, Bm, bias, R, Cm, M, N, K,
                  lda, ldb, ldc, ldr, sA, sB, sC, sBias, sR); break;
      case 2: gemm_kernel<2><<<grid, 256, 0, stream>>>(A, Bm, bias, R, Cm, M, N, K,
                  lda, ldb, ldc, ldr, sA, sB, sC, sBias, sR); break;
      case 3: gemm_kernel<3><<<grid, 256, 0, stream>>>(A, Bm, bias, R, Cm, M, N, K,
                  lda, ldb, ldc, ldr, sA, sB, sC, sBias, sR); break;
      default: gemm_kernel<0><<<grid, 256, 0, stream>>>(A, Bm, bias, R, Cm, M, N, K,
                  lda, ldb, ldc, ldr, sA, sB, sC, sBias, sR); break;
    }
  };
  auto GEMM1 = [&](const float* A, const float* Bm, const float* bias, const float* R,
                   float* Cm, int M, int N, int K, int act) {
    GEMM(A, Bm, bias, R, Cm, M, N, K, K, N, N, N, 0, 0, 0, 0, 0, 1, act);
  };

  hipMemsetAsync(out_aux, 0, sizeof(float), stream);

  // ---- text encoder: mlp_enc(text_embedding) -> out_text ----
  addln_kernel<<<NB, 256, 0, stream>>>(text_emb, nullptr, P[0], P[1], RB, D_PRE);
  GEMM1(RB, P[2], P[3], nullptr, RC, NB, D_PRE, D_PRE, 1);       // leaky
  GEMM1(RC, P[4], P[5], nullptr, out_text, NB, DMODEL, D_PRE, 0);

  // ---- token encoder: mlp_enc(token_embedding) -> RD (tok) ----
  addln_kernel<<<NTOK, 256, 0, stream>>>(token_emb, nullptr, P[6], P[7], RB, D_PRE);
  GEMM1(RB, P[8], P[9], nullptr, RF, NTOK, D_PRE, D_PRE, 1);
  GEMM1(RF, P[10], P[11], nullptr, RD, NTOK, DMODEL, D_PRE, 0);

  // ---- moe_block (shared host sequence) ----
  auto MOE_BLOCK = [&](const float* xin, float* xwork, int N, int cap,
                       const float* const* p, float* finalOut) {
    // ffb: X1 = xin + gelu(LN(xin)@w1+b1)@w2+b2
    addln_kernel<<<N, 256, 0, stream>>>(xin, nullptr, p[0], p[1], RC, DMODEL);
    GEMM1(RC, p[2], p[3], nullptr, RB, N, DFF, DMODEL, 2);
    GEMM1(RB, p[4], p[5], xin, xwork, N, DMODEL, DFF, 0);
    // gate + routing
    GEMM1(xwork, p[12], nullptr, nullptr, LG, N, NEXP, DMODEL, 0);
    hipMemsetAsync(LA, 0, (NEXP + 1) * sizeof(float), stream);
    route_kernel<<<(N + 255) / 256, 256, 0, stream>>>(LG, GV, GI, KP, LA, N);
    scan_kernel<<<1, 256, 0, stream>>>(GI, KP, PS, N * 2);
    hipMemsetAsync(RE, 0, (size_t)NEXP * cap * DMODEL * sizeof(float), stream);
    scatter_kernel<<<N * 2, 256, 0, stream>>>(xwork, GI, KP, PS, RE, cap);
    // experts: buf @ ew1 (+eb1) -> RF [cap x 4096], GLU, @ ew2 (+eb2) -> RG
    GEMM(RE, p[13], p[14], nullptr, RF, cap, 2 * DFF, DMODEL,
         DMODEL, 2 * DFF, 2 * DFF, 0,
         (long long)cap * DMODEL, (long long)DMODEL * 2 * DFF,
         (long long)cap * 2 * DFF, 2 * DFF, 0, NEXP, 0);
    {
      long long tot = (long long)NEXP * cap * DFF;
      glu_kernel<<<(unsigned)((tot + 255) / 256), 256, 0, stream>>>(RF, (long long)NEXP * cap);
    }
    GEMM(RF, p[15], p[16], nullptr, RG, cap, DMODEL, DFF,
         2 * DFF, DMODEL, DMODEL, 0,
         (long long)cap * 2 * DFF, (long long)DFF * DMODEL,
         (long long)cap * DMODEL, DMODEL, 0, NEXP, 0);
    combine_kernel<<<N, 256, 0, stream>>>(xwork, RG, GV, GI, KP, PS, cap);
    aux_finalize_kernel<<<1, 32, 0, stream>>>(LA, out_aux);
    // ffa
    addln_kernel<<<N, 256, 0, stream>>>(xwork, nullptr, p[6], p[7], RC, DMODEL);
    GEMM1(RC, p[8], p[9], nullptr, RB, N, DFF, DMODEL, 2);
    GEMM1(RB, p[10], p[11], xwork, finalOut, N, DMODEL, DFF, 0);
  };

  // ---- tokmoe: tok_out = moe_block(tok) ----
  MOE_BLOCK(RD, RA, NTOK, CAP_TOK, P + 12, out_tok);

  // ---- concat(tok, geo_emb) * sqrt(D) + PE -> RA ----
  concat_pe_kernel<<<(NCAT * DMODEL + 255) / 256, 256, 0, stream>>>(RD, P[94], geo_code, RA);

  // ---- transformer layers ----
  for (int i = 0; i < 4; ++i) {
    const float* const* tp = P + 46 + i * 12;
    GEMM1(RA, tp[0], tp[1], nullptr, RB, NCAT, 3 * DMODEL, DMODEL, 0);   // qkv
    attn_kernel<<<NB * NHEAD, 128, 0, stream>>>(RB, token_mask, RC);
    GEMM1(RC, tp[2], tp[3], nullptr, RB, NCAT, DMODEL, DMODEL, 0);       // proj
    addln_kernel<<<NCAT, 256, 0, stream>>>(RA, RB, tp[4], tp[5], RA, DMODEL);
    GEMM1(RA, tp[8], tp[9], nullptr, RB, NCAT, DFF, DMODEL, 3);          // relu
    GEMM1(RB, tp[10], tp[11], nullptr, RC, NCAT, DMODEL, DFF, 0);
    addln_kernel<<<NCAT, 256, 0, stream>>>(RA, RC, tp[6], tp[7], RA, DMODEL);
  }

  // ---- geomoe -> RC, slice -> out_geo ----
  MOE_BLOCK(RA, RA, NCAT, CAP_GEO, P + 29, RC);
  slice_geo_kernel<<<(NB * SGEO * DMODEL + 255) / 256, 256, 0, stream>>>(RC, out_geo);
}
